// GraphNeuralNetwork_36842229465912
// MI455X (gfx1250) — compile-verified, matrix operand order checked by reading, not run on hardware
//
#include <hip/hip_runtime.h>
#include <hip/hip_bf16.h>

// ============================================================================
// CDNA5 (gfx1250, wave32) implementation of the bipartite GNN.
// All 64-wide GEMMs use v_wmma_f32_16x16x32_bf16 (fp32 accumulate).
// Weights live in LDS pre-swizzled fragment-major so each WMMA B operand is
// exactly two ds_load_b128. Staging uses float4 loads + packed bf16x4 LDS
// stores; gathers/scatters use 32-bit byte offsets (SADDR addressing).
// Workspace requirement: ~116 MB.
// ============================================================================

#define EPS 1e-5f

typedef __attribute__((ext_vector_type(16))) __bf16 v16bf;
typedef __attribute__((ext_vector_type(8)))  __bf16 v8bf;
typedef __attribute__((ext_vector_type(4)))  __bf16 v4bf;
typedef __attribute__((ext_vector_type(8)))  float  v8f;

// CDNA5 split-counter wait: all LDS writes by this wave visible before reads.
__device__ inline void lds_fence() { asm volatile("s_wait_dscnt 0" ::: "memory"); }

__device__ inline v8f wmma_bf16(v16bf a, v16bf b, v8f c) {
  // 8 args: (neg_a, A, neg_b, B, c_mod, C, reuse_a, reuse_b)
  return __builtin_amdgcn_wmma_f32_16x16x32_bf16(false, a, false, b, (short)0, c,
                                                 false, false);
}

// A fragment: 16xK row-major bf16 tile in LDS with row stride `ld` (bf16).
// 16-bit A layout (16x32): lanes 0-15, rows M=0..15, hold K=kb..kb+7 and
// kb+16..kb+23 with kb=kc*32; lanes 16-31 same rows with kb=kc*32+8.
// Both 8-element runs are contiguous & 16B-aligned -> two ds_load_b128.
__device__ inline v16bf a_frag(const __bf16* A, int ld, int lane, int kc) {
  int row = lane & 15;
  int kb  = kc * 32 + ((lane >> 4) << 3);
  const __bf16* p = A + row * ld + kb;
  v8bf lo = *(const v8bf*)p;
  v8bf hi = *(const v8bf*)(p + 16);
  return __builtin_shufflevector(lo, hi, 0, 1, 2, 3, 4, 5, 6, 7,
                                 8, 9, 10, 11, 12, 13, 14, 15);
}

// ---------------------------------------------------------------------------
// Swizzled B-weight staging: W is KxN (N=64, K=64 or 128) row-major in global.
// LDS layout is fragment-major: frag (kc*4+nc), then lane, then 16 contiguous
// bf16 -> b_frag_s is a single 32B contiguous read (two ds_load_b128).
// ---------------------------------------------------------------------------
__device__ inline void stage_b_swizzled(const float* __restrict__ W, __bf16* sWs,
                                        int K, int tid, int nthreads) {
  int total = K * 64;
  for (int d = tid; d < total; d += nthreads) {
    int j    = d & 15;
    int lane = (d >> 4) & 31;
    int frag = d >> 9;                 // kc*4 + nc
    int kc   = frag >> 2, nc = frag & 3;
    int col  = nc * 16 + (lane & 15);
    int kb   = kc * 32 + ((lane >> 4) << 3);
    int k    = kb + ((j < 8) ? j : (j + 8));
    sWs[d] = (__bf16)W[k * 64 + col];
  }
}

__device__ inline v16bf b_frag_s(const __bf16* sWs, int lane, int kc, int nc) {
  return *(const v16bf*)(sWs + ((((kc << 2) + nc) << 5) + lane) * 16);
}

// 32-bit byte-offset load/atomic helpers (SADDR + VADDR32 addressing).
__device__ inline const float4* f4p(const float* base, unsigned boff) {
  return (const float4*)((const char*)base + (size_t)boff);
}
__device__ inline float ldf(const float* base, unsigned boff) {
  return *(const float*)((const char*)base + (size_t)boff);
}

// ---------------------------------------------------------------------------
// Per-column sum / sum-of-squares for BatchNorm (C <= 64). stats = [sum[C], sq[C]]
// ---------------------------------------------------------------------------
__global__ __launch_bounds__(256)
void colstats_kernel(const float* __restrict__ x, int n, int C,
                     float* __restrict__ stats) {
  __shared__ float ssum[64], ssq[64];
  int tid = threadIdx.x;
  if (tid < 64) { ssum[tid] = 0.f; ssq[tid] = 0.f; }
  __syncthreads();
  int c   = tid % C;
  int rg  = tid / C;
  int nrg = blockDim.x / C;
  float s = 0.f, s2 = 0.f;
  if (rg < nrg) {
    for (long r = (long)blockIdx.x * nrg + rg; r < n; r += (long)gridDim.x * nrg) {
      float v = x[r * C + c];
      s += v; s2 += v * v;
    }
  }
  atomicAdd(&ssum[c], s); atomicAdd(&ssq[c], s2);
  __syncthreads();
  if (tid < C) { atomicAdd(&stats[tid], ssum[tid]); atomicAdd(&stats[C + tid], ssq[tid]); }
}

// ---------------------------------------------------------------------------
// Embedding: BN(x) -> [NF->64] relu (scalar, tiny K) -> [64->64] relu (WMMA)
// ---------------------------------------------------------------------------
template <int NF>
__global__ __launch_bounds__(256)
void embed_kernel(const float* __restrict__ x, const float* __restrict__ stats,
                  const float* __restrict__ bn_g, const float* __restrict__ bn_b,
                  const float* __restrict__ w1, const float* __restrict__ b1,
                  const float* __restrict__ w2, const float* __restrict__ b2,
                  float* __restrict__ out, int n) {
  constexpr int WAVES = 8;
  __shared__ __align__(32) __bf16 sW2[64 * 64];
  __shared__ float  sW1[NF * 64];
  __shared__ float  sX[WAVES][16][NF];
  __shared__ __align__(16) __bf16 sH[WAVES][16][64];
  __shared__ float  sScale[NF], sBias[NF];
  int tid = threadIdx.x, w = tid >> 5, lane = tid & 31;
  stage_b_swizzled(w2, sW2, 64, tid, 256);
  for (int i = tid; i < NF * 64; i += 256) sW1[i] = w1[i];
  if (tid < NF) {
    float mean = stats[tid] / n;
    float var  = stats[NF + tid] / n - mean * mean;
    float sc   = bn_g[tid] * rsqrtf(var + EPS);
    sScale[tid] = sc;
    sBias[tid]  = bn_b[tid] - mean * sc;
  }
  __syncthreads();
  // hoisted per-lane constants
  float b1c0 = b1[lane], b1c1 = b1[lane + 32];
  int   colD = lane & 15;
  float b2c[4];
#pragma unroll
  for (int nc = 0; nc < 4; ++nc) b2c[nc] = b2[nc * 16 + colD];
  int tiles = n >> 4;
  for (int t = blockIdx.x * WAVES + w; t < tiles; t += gridDim.x * WAVES) {
    long r0 = (long)t << 4;
    for (int i = lane; i < 16 * NF; i += 32) {
      int r = i / NF, c = i % NF;
      sX[w][r][c] = x[(r0 + r) * NF + c] * sScale[c] + sBias[c];
    }
    lds_fence();
#pragma unroll
    for (int half = 0; half < 2; ++half) {        // each lane: 2 cols x 16 rows
      int col = lane + half * 32;
      float b1c = half ? b1c1 : b1c0;
#pragma unroll
      for (int r = 0; r < 16; ++r) {
        float acc = b1c;
#pragma unroll
        for (int k = 0; k < NF; ++k) acc += sX[w][r][k] * sW1[k * 64 + col];
        sH[w][r][col] = (__bf16)fmaxf(acc, 0.f);
      }
    }
    lds_fence();
    v16bf a0 = a_frag(&sH[w][0][0], 64, lane, 0);
    v16bf a1 = a_frag(&sH[w][0][0], 64, lane, 1);
#pragma unroll
    for (int nc = 0; nc < 4; ++nc) {
      v8f acc = {};
      acc = wmma_bf16(a0, b_frag_s(sW2, lane, 0, nc), acc);
      acc = wmma_bf16(a1, b_frag_s(sW2, lane, 1, nc), acc);
      int col = nc * 16 + colD;
      int mb  = (lane >> 4) << 3;
#pragma unroll
      for (int r = 0; r < 8; ++r)
        out[(r0 + mb + r) * 64 + col] = fmaxf(acc[r] + b2c[nc], 0.f);
    }
  }
}

// ---------------------------------------------------------------------------
// Y[n,64] = X[n,64] @ W[64,64] (+bias)(+relu) — WMMA, one wave per 16-row tile
// ---------------------------------------------------------------------------
template <int RELU, int BIAS>
__global__ __launch_bounds__(256)
void gemm64_kernel(const float* __restrict__ X, const float* __restrict__ W,
                   const float* __restrict__ bias, float* __restrict__ Y, int n) {
  constexpr int WAVES = 8;
  __shared__ __align__(32) __bf16 sW[4096];
  __shared__ __align__(16) __bf16 sA[WAVES][16][64];
  int tid = threadIdx.x, w = tid >> 5, lane = tid & 31;
  stage_b_swizzled(W, sW, 64, tid, 256);
  __syncthreads();
  int   colD = lane & 15;
  float bc[4] = {0.f, 0.f, 0.f, 0.f};
  if (BIAS) {
#pragma unroll
    for (int nc = 0; nc < 4; ++nc) bc[nc] = bias[nc * 16 + colD];
  }
  const float4* Xv = (const float4*)X;
  int tiles = n >> 4;
  for (int t = blockIdx.x * WAVES + w; t < tiles; t += gridDim.x * WAVES) {
    long r0 = (long)t << 4;
#pragma unroll
    for (int it = 0; it < 8; ++it) {             // 256 float4 per tile
      int i = lane + (it << 5);
      float4 v = Xv[(size_t)t * 256 + i];
      v4bf o; o[0] = (__bf16)v.x; o[1] = (__bf16)v.y;
              o[2] = (__bf16)v.z; o[3] = (__bf16)v.w;
      *(v4bf*)((&sA[w][0][0]) + (i << 2)) = o;
    }
    lds_fence();
    v16bf a0 = a_frag(&sA[w][0][0], 64, lane, 0);
    v16bf a1 = a_frag(&sA[w][0][0], 64, lane, 1);
#pragma unroll
    for (int nc = 0; nc < 4; ++nc) {
      v8f acc = {};
      acc = wmma_bf16(a0, b_frag_s(sW, lane, 0, nc), acc);
      acc = wmma_bf16(a1, b_frag_s(sW, lane, 1, nc), acc);
      int col = nc * 16 + colD;
      int mb  = (lane >> 4) << 3;
#pragma unroll
      for (int r = 0; r < 8; ++r) {
        float y = acc[r] + bc[nc];
        if (RELU) y = fmaxf(y, 0.f);
        Y[(r0 + mb + r) * 64 + col] = y;
      }
    }
  }
}

// ---------------------------------------------------------------------------
// Pass A of edge conv: column stats of
//   joint = (ef*es+et)*we + preL[ci] + preR[vi]   (edge BN folded into we)
// ---------------------------------------------------------------------------
__global__ __launch_bounds__(256)
void edge_stats_kernel(const float* __restrict__ ef, const float* __restrict__ we,
                       const float* __restrict__ eStats, const float* __restrict__ eg,
                       const float* __restrict__ eb,
                       const float* __restrict__ preL, const float* __restrict__ preR,
                       const int* __restrict__ idxL, const int* __restrict__ idxR,
                       int nE, float* __restrict__ stats) {
  __shared__ float ssum[64], ssq[64];
  int tid = threadIdx.x;
  if (tid < 64) { ssum[tid] = 0.f; ssq[tid] = 0.f; }
  __syncthreads();
  int c = tid & 63, g = tid >> 6;
  float em  = eStats[0] / nE;
  float ev  = eStats[1] / nE - em * em;
  float esc = eg[0] * rsqrtf(ev + EPS);
  float ebi = eb[0] - em * esc;
  float wec = we[c];
  float wes = esc * wec, web = ebi * wec;
  unsigned cb = (unsigned)c << 2;
  float s = 0.f, s2 = 0.f;
  for (long e = (long)blockIdx.x * 4 + g; e < nE; e += (long)gridDim.x * 4) {
    unsigned il = (unsigned)idxL[e] << 8;
    unsigned ir = (unsigned)idxR[e] << 8;
    float j = ef[e] * wes + web + ldf(preL, il + cb) + ldf(preR, ir + cb);
    s += j; s2 += j * j;
  }
  atomicAdd(&ssum[c], s); atomicAdd(&ssq[c], s2);
  __syncthreads();
  if (tid < 64) { atomicAdd(&stats[tid], ssum[tid]); atomicAdd(&stats[64 + tid], ssq[tid]); }
}

// ---------------------------------------------------------------------------
// Pass B: recompute joint, BN+ReLU, WMMA @ fw (+fb), scatter-add into out.
// Each lane owns 4 fixed columns in staging -> all BN constants in registers.
// ---------------------------------------------------------------------------
__global__ __launch_bounds__(256)
void edge_conv_kernel(const float* __restrict__ ef, const float* __restrict__ we,
                      const float* __restrict__ eStats, const float* __restrict__ eg,
                      const float* __restrict__ eb,
                      const float* __restrict__ preL, const float* __restrict__ preR,
                      const int* __restrict__ idxL, const int* __restrict__ idxR,
                      const int* __restrict__ sidx, int nE,
                      const float* __restrict__ stats,
                      const float* __restrict__ fg, const float* __restrict__ fbn,
                      const float* __restrict__ fw, const float* __restrict__ fb,
                      float* __restrict__ out) {
  constexpr int WAVES = 8;
  __shared__ __align__(32) __bf16 sFW[4096];
  __shared__ __align__(16) __bf16 sJ[WAVES][16][64];
  __shared__ float sWeS[64], sWeB[64], sScale[64], sBias[64];
  int tid = threadIdx.x, w = tid >> 5, lane = tid & 31;
  stage_b_swizzled(fw, sFW, 64, tid, 256);
  if (tid < 64) {
    float em  = eStats[0] / nE;
    float ev  = eStats[1] / nE - em * em;
    float esc = eg[0] * rsqrtf(ev + EPS);
    float ebi = eb[0] - em * esc;
    float wv  = we[tid];
    sWeS[tid] = esc * wv; sWeB[tid] = ebi * wv;
    float mean = stats[tid] / nE;
    float var  = stats[64 + tid] / nE - mean * mean;
    float sc   = fg[tid] * rsqrtf(var + EPS);
    sScale[tid] = sc;
    sBias[tid]  = fbn[tid] - mean * sc;
  }
  __syncthreads();
  int cl = (lane & 15) << 2;                    // this lane's 4 staging columns
  float ws4[4], wb4[4], sc4[4], bi4[4];
#pragma unroll
  for (int k = 0; k < 4; ++k) {
    ws4[k] = sWeS[cl + k]; wb4[k] = sWeB[cl + k];
    sc4[k] = sScale[cl + k]; bi4[k] = sBias[cl + k];
  }
  int   colD = lane & 15;
  float fb4[4];
#pragma unroll
  for (int nc = 0; nc < 4; ++nc) fb4[nc] = fb[nc * 16 + colD];
  unsigned clB = (unsigned)cl << 2;             // byte offset of lane's columns
  int tiles = nE >> 4;
  for (int t = blockIdx.x * WAVES + w; t < tiles; t += gridDim.x * WAVES) {
    long e0 = (long)t << 4;
    // 16 edge records loaded once, distributed by shuffle
    int   eL  = (int)e0 + colD;
    int   myL = idxL[eL], myR = idxR[eL], myS = sidx[eL];
    float myE = ef[eL];
#pragma unroll
    for (int it = 0; it < 8; ++it) {            // rows: lanes 0-15 even, 16-31 odd
      int r = (lane >> 4) + (it << 1);
      unsigned il = (unsigned)__shfl(myL, r, 32) << 8;
      unsigned ir = (unsigned)__shfl(myR, r, 32) << 8;
      float    ev = __shfl(myE, r, 32);
      float4 pl = *f4p(preL, il + clB);
      float4 pr = *f4p(preR, ir + clB);
      v4bf o;
      o[0] = (__bf16)fmaxf((ev * ws4[0] + wb4[0] + pl.x + pr.x) * sc4[0] + bi4[0], 0.f);
      o[1] = (__bf16)fmaxf((ev * ws4[1] + wb4[1] + pl.y + pr.y) * sc4[1] + bi4[1], 0.f);
      o[2] = (__bf16)fmaxf((ev * ws4[2] + wb4[2] + pl.z + pr.z) * sc4[2] + bi4[2], 0.f);
      o[3] = (__bf16)fmaxf((ev * ws4[3] + wb4[3] + pl.w + pr.w) * sc4[3] + bi4[3], 0.f);
      *(v4bf*)(&sJ[w][r][cl]) = o;
    }
    lds_fence();
    int mb = (lane >> 4) << 3;
    unsigned rowB[8];
#pragma unroll
    for (int r = 0; r < 8; ++r) rowB[r] = (unsigned)__shfl(myS, mb + r, 32) << 8;
    v16bf a0 = a_frag(&sJ[w][0][0], 64, lane, 0);
    v16bf a1 = a_frag(&sJ[w][0][0], 64, lane, 1);
#pragma unroll
    for (int nc = 0; nc < 4; ++nc) {
      v8f acc = {};
      acc = wmma_bf16(a0, b_frag_s(sFW, lane, 0, nc), acc);
      acc = wmma_bf16(a1, b_frag_s(sFW, lane, 1, nc), acc);
      unsigned cb = (unsigned)(nc * 16 + colD) << 2;
#pragma unroll
      for (int r = 0; r < 8; ++r)
        atomicAdd((float*)((char*)out + (size_t)(rowB[r] + cb)), acc[r] + fb4[nc]);
    }
  }
}

// ---------------------------------------------------------------------------
// out = relu(relu([BN(conv), prev] @ ow1 + ob1) @ ow2 + ob2)   (in-place safe)
// ---------------------------------------------------------------------------
__global__ __launch_bounds__(128)
void post_conv_kernel(const float* __restrict__ conv, const float* __restrict__ stats,
                      const float* __restrict__ pg, const float* __restrict__ pb,
                      const float* __restrict__ prev,
                      const float* __restrict__ ow1, const float* __restrict__ ob1,
                      const float* __restrict__ ow2, const float* __restrict__ ob2,
                      float* __restrict__ out, int n) {
  constexpr int WAVES = 4;
  __shared__ __align__(32) __bf16 sW1[128 * 64];
  __shared__ __align__(32) __bf16 sW2[64 * 64];
  __shared__ __align__(16) __bf16 sA[WAVES][16][128];
  __shared__ __align__(16) __bf16 sH[WAVES][16][64];
  __shared__ float  sScale[64], sBias[64];
  int tid = threadIdx.x, w = tid >> 5, lane = tid & 31;
  stage_b_swizzled(ow1, sW1, 128, tid, 128);
  stage_b_swizzled(ow2, sW2, 64, tid, 128);
  if (tid < 64) {
    float mean = stats[tid] / n;
    float var  = stats[64 + tid] / n - mean * mean;
    float sc   = pg[tid] * rsqrtf(var + EPS);
    sScale[tid] = sc;
    sBias[tid]  = pb[tid] - mean * sc;
  }
  __syncthreads();
  // staging: lanes 0-15 -> conv (with BN), lanes 16-31 -> prev; 4 cols per lane
  int side = lane >> 4;
  int cq   = lane & 15;                 // quad index within the 64-col source row
  float sc4[4] = {0, 0, 0, 0}, bi4[4] = {0, 0, 0, 0};
  if (!side) {
#pragma unroll
    for (int k = 0; k < 4; ++k) { sc4[k] = sScale[cq * 4 + k]; bi4[k] = sBias[cq * 4 + k]; }
  }
  const float4* srcv = side ? (const float4*)prev : (const float4*)conv;
  int   colD = lane & 15;
  float ob1c[4], ob2c[4];
#pragma unroll
  for (int nc = 0; nc < 4; ++nc) { ob1c[nc] = ob1[nc * 16 + colD]; ob2c[nc] = ob2[nc * 16 + colD]; }
  int tiles = n >> 4;
  for (int t = blockIdx.x * WAVES + w; t < tiles; t += gridDim.x * WAVES) {
    long r0 = (long)t << 4;
#pragma unroll
    for (int r = 0; r < 16; ++r) {
      float4 v = srcv[(r0 + r) * 16 + cq];
      if (!side) {
        v.x = v.x * sc4[0] + bi4[0]; v.y = v.y * sc4[1] + bi4[1];
        v.z = v.z * sc4[2] + bi4[2]; v.w = v.w * sc4[3] + bi4[3];
      }
      v4bf o; o[0] = (__bf16)v.x; o[1] = (__bf16)v.y;
              o[2] = (__bf16)v.z; o[3] = (__bf16)v.w;
      *(v4bf*)(&sA[w][r][lane << 2]) = o;
    }
    lds_fence();
    v16bf af[4];
#pragma unroll
    for (int kc = 0; kc < 4; ++kc) af[kc] = a_frag(&sA[w][0][0], 128, lane, kc);
#pragma unroll
    for (int nc = 0; nc < 4; ++nc) {
      v8f acc = {};
#pragma unroll
      for (int kc = 0; kc < 4; ++kc)
        acc = wmma_bf16(af[kc], b_frag_s(sW1, lane, kc, nc), acc);
      int col = nc * 16 + colD;
      int mb  = (lane >> 4) << 3;
#pragma unroll
      for (int r = 0; r < 8; ++r)
        sH[w][mb + r][col] = (__bf16)fmaxf(acc[r] + ob1c[nc], 0.f);
    }
    lds_fence();
    v16bf h0 = a_frag(&sH[w][0][0], 64, lane, 0);
    v16bf h1 = a_frag(&sH[w][0][0], 64, lane, 1);
#pragma unroll
    for (int nc = 0; nc < 4; ++nc) {
      v8f acc = {};
      acc = wmma_bf16(h0, b_frag_s(sW2, lane, 0, nc), acc);
      acc = wmma_bf16(h1, b_frag_s(sW2, lane, 1, nc), acc);
      int col = nc * 16 + colD;
      int mb  = (lane >> 4) << 3;
#pragma unroll
      for (int r = 0; r < 8; ++r)
        out[(r0 + mb + r) * 64 + col] = fmaxf(acc[r] + ob2c[nc], 0.f);
    }
  }
}

// ---------------------------------------------------------------------------
// Final: out[i] = dot(h[cand[i]], w2) + b2   — one wave32 per candidate
// ---------------------------------------------------------------------------
__global__ void cand_head_kernel(const float* __restrict__ h, const float* __restrict__ w2,
                                 const float* __restrict__ b2, const int* __restrict__ cand,
                                 float* __restrict__ out, int n) {
  int gw   = (int)((blockIdx.x * (long)blockDim.x + threadIdx.x) >> 5);
  int lane = threadIdx.x & 31;
  if (gw >= n) return;
  unsigned rb = (unsigned)cand[gw] << 8;
  float s = ldf(h, rb + (lane << 2)) * w2[lane] +
            ldf(h, rb + ((lane + 32) << 2)) * w2[lane + 32];
#pragma unroll
  for (int off = 16; off > 0; off >>= 1) s += __shfl_down(s, off, 32);
  if (lane == 0) out[gw] = s + b2[0];
}

// ===========================================================================
// Host side. Params arrive jax-tree-flattened (dict keys sorted recursively):
//   top: cons_emb, conv_c2v, conv_v2c, edge_bn_b, edge_bn_g,
//        out_b1, out_b2, out_w1, out_w2, var_emb
//   emb: b1,b2,bn_b,bn_g,w1,w2
//   conv: bl,fb,fin_bn_b,fin_bn_g,fw,ob1,ob2,ow1,ow2,post_bn_b,post_bn_g,we,wl,wr
// ===========================================================================
extern "C" void kernel_launch(void* const* d_in, const int* in_sizes, int n_in,
                              void* d_out, int out_size, void* d_ws, size_t ws_size,
                              hipStream_t stream) {
  const float* cons_x  = (const float*)d_in[0];
  const int*   ei_cons = (const int*)d_in[1];
  const int*   ei_vars = (const int*)d_in[2];
  const float* edge_x  = (const float*)d_in[3];
  const float* var_x   = (const float*)d_in[4];
  const int*   cand    = (const int*)d_in[5];
  const float* const* P = (const float* const*)(d_in + 6);

  // emb offsets within P:  0:b1 1:b2 2:bn_b 3:bn_g 4:w1 5:w2
  const int C2V = 6, V2C = 20;  // conv_c2v sorts before conv_v2c
  // conv offsets: +0:bl +1:fb +2:fin_bn_b +3:fin_bn_g +4:fw +5:ob1 +6:ob2
  //               +7:ow1 +8:ow2 +9:post_bn_b +10:post_bn_g +11:we +12:wl +13:wr
  const int EDGE_B = 34, EDGE_G = 35;
  const int OUT_B1 = 36, OUT_B2 = 37, OUT_W1 = 38, OUT_W2 = 39;
  const int VEMB = 40;  // var_emb: +0..5 as emb

  int nCons  = in_sizes[0] / 5;
  int nEdges = in_sizes[1];
  int nVars  = in_sizes[4] / 19;

  float* ws = (float*)d_ws;
  size_t off = 0;
  auto alloc = [&](size_t cnt) { float* p = ws + off; off += cnt; return p; };
  float* stCons  = alloc(16);          // 2*5 (padded)
  float* stVar   = alloc(64);          // 2*19 (padded)
  float* stEdge  = alloc(16);          // 2*1 (padded)
  float* stFinA  = alloc(128);
  float* stFinB  = alloc(128);
  float* stPostA = alloc(128);
  float* stPostB = alloc(128);
  size_t statFloats = off;             // contiguous stats region
  float* cf    = alloc((size_t)nCons * 64);
  float* vf    = alloc((size_t)nVars * 64);
  float* preL  = alloc((size_t)nCons * 64);
  float* preR  = alloc((size_t)nVars * 64);
  float* convC = alloc((size_t)nCons * 64);
  float* convV = alloc((size_t)nVars * 64);
  float* hbuf  = preR;                 // reused after both convs are done

  // Zero accumulators (poisoned workspace + scatter targets) every call.
  hipMemsetAsync(ws, 0, statFloats * sizeof(float), stream);
  hipMemsetAsync(convC, 0, (size_t)nCons * 64 * sizeof(float), stream);
  hipMemsetAsync(convV, 0, (size_t)nVars * 64 * sizeof(float), stream);

  int tilesC = nCons >> 4, tilesV = nVars >> 4, tilesE = nEdges >> 4;
  auto gb8 = [](int t) { return (t + 7) / 8; };
  auto gb4 = [](int t) { return (t + 3) / 4; };

  // --- BatchNorm input statistics ---
  colstats_kernel<<<256, 256, 0, stream>>>(cons_x, nCons, 5, stCons);
  colstats_kernel<<<512, 256, 0, stream>>>(var_x, nVars, 19, stVar);
  colstats_kernel<<<512, 256, 0, stream>>>(edge_x, nEdges, 1, stEdge);

  // --- Embeddings ---
  embed_kernel<5><<<gb8(tilesC), 256, 0, stream>>>(
      cons_x, stCons, P[3], P[2], P[4], P[0], P[5], P[1], cf, nCons);
  embed_kernel<19><<<gb8(tilesV), 256, 0, stream>>>(
      var_x, stVar, P[VEMB + 3], P[VEMB + 2], P[VEMB + 4], P[VEMB + 0],
      P[VEMB + 5], P[VEMB + 1], vf, nVars);

  // --- Conv 1: vars -> cons (scatter over ei_cons, prev = cf) ---
  gemm64_kernel<0, 1><<<gb8(tilesC), 256, 0, stream>>>(cf, P[V2C + 12], P[V2C + 0], preL, nCons);
  gemm64_kernel<0, 0><<<gb8(tilesV), 256, 0, stream>>>(vf, P[V2C + 13], nullptr, preR, nVars);
  edge_stats_kernel<<<2048, 256, 0, stream>>>(
      edge_x, P[V2C + 11], stEdge, P[EDGE_G], P[EDGE_B],
      preL, preR, ei_cons, ei_vars, nEdges, stFinA);
  edge_conv_kernel<<<gb8(tilesE), 256, 0, stream>>>(
      edge_x, P[V2C + 11], stEdge, P[EDGE_G], P[EDGE_B],
      preL, preR, ei_cons, ei_vars, /*scatter=*/ei_cons, nEdges,
      stFinA, P[V2C + 3], P[V2C + 2], P[V2C + 4], P[V2C + 1], convC);
  colstats_kernel<<<256, 256, 0, stream>>>(convC, nCons, 64, stPostA);
  post_conv_kernel<<<gb4(tilesC), 128, 0, stream>>>(
      convC, stPostA, P[V2C + 10], P[V2C + 9], /*prev=*/cf,
      P[V2C + 7], P[V2C + 5], P[V2C + 8], P[V2C + 6], /*out in place=*/cf, nCons);

  // --- Conv 2: cons -> vars (scatter over ei_vars, prev = vf) ---
  gemm64_kernel<0, 1><<<gb8(tilesC), 256, 0, stream>>>(cf, P[C2V + 12], P[C2V + 0], preL, nCons);
  gemm64_kernel<0, 0><<<gb8(tilesV), 256, 0, stream>>>(vf, P[C2V + 13], nullptr, preR, nVars);
  edge_stats_kernel<<<2048, 256, 0, stream>>>(
      edge_x, P[C2V + 11], stEdge, P[EDGE_G], P[EDGE_B],
      preL, preR, ei_cons, ei_vars, nEdges, stFinB);
  edge_conv_kernel<<<gb8(tilesE), 256, 0, stream>>>(
      edge_x, P[C2V + 11], stEdge, P[EDGE_G], P[EDGE_B],
      preL, preR, ei_cons, ei_vars, /*scatter=*/ei_vars, nEdges,
      stFinB, P[C2V + 3], P[C2V + 2], P[C2V + 4], P[C2V + 1], convV);
  colstats_kernel<<<512, 256, 0, stream>>>(convV, nVars, 64, stPostB);
  post_conv_kernel<<<gb4(tilesV), 128, 0, stream>>>(
      convV, stPostB, P[C2V + 10], P[C2V + 9], /*prev=*/vf,
      P[C2V + 7], P[C2V + 5], P[C2V + 8], P[C2V + 6], /*out in place=*/vf, nVars);

  // --- Output head + candidate gather ---
  gemm64_kernel<1, 1><<<gb8(tilesV), 256, 0, stream>>>(vf, P[OUT_W1], P[OUT_B1], hbuf, nVars);
  cand_head_kernel<<<(out_size * 32 + 255) / 256, 256, 0, stream>>>(
      hbuf, P[OUT_W2], P[OUT_B2], cand, (float*)d_out, out_size);
}
